// PytorchLTX2Attention_30605936952017
// MI455X (gfx1250) — matrix operand order repaired and checked by reading, not compiled, not run on hardware
//
#include <hip/hip_runtime.h>

// ---------------------------------------------------------------------------
// Problem constants (match reference)
// ---------------------------------------------------------------------------
#define B_     2
#define S_     2048
#define D_     2048
#define H_     32
#define DH_    64
#define INNER_ 2048
#define EPS_   1e-6f

typedef __bf16 bf16;
typedef __attribute__((ext_vector_type(4)))  __bf16 v4bf;
typedef __attribute__((ext_vector_type(8)))  __bf16 v8bf;
typedef __attribute__((ext_vector_type(16))) __bf16 v16bf;
typedef __attribute__((ext_vector_type(8)))  float  v8f;

// ---------------------------------------------------------------------------
// CDNA5 async global->LDS copy (ASYNCcnt-tracked DMA), with sync fallback
// ---------------------------------------------------------------------------
#if defined(__has_builtin)
#if __has_builtin(__builtin_amdgcn_global_load_async_to_lds_b128)
#define HAVE_ASYNC 1
#endif
#endif
#ifndef HAVE_ASYNC
#define HAVE_ASYNC 0
#endif

typedef int vi4 __attribute__((vector_size(16)));           // 128-bit payload
typedef __attribute__((address_space(1))) vi4 gvi4;          // global
typedef __attribute__((address_space(3))) vi4 svi4;          // LDS

__device__ __forceinline__ void cp16(bf16* dst_lds, const bf16* src_global) {
#if HAVE_ASYNC
    __builtin_amdgcn_global_load_async_to_lds_b128(
        (gvi4*)(void*)const_cast<bf16*>(src_global),
        (svi4*)(void*)dst_lds,
        0, 0);
#else
    *reinterpret_cast<float4*>(dst_lds) =
        *reinterpret_cast<const float4*>(src_global);
#endif
}

__device__ __forceinline__ void wait_async0() {
#if HAVE_ASYNC
#if __has_builtin(__builtin_amdgcn_s_wait_asynccnt)
    __builtin_amdgcn_s_wait_asynccnt(0);
#else
    asm volatile("s_wait_asynccnt 0x0" ::: "memory");
#endif
#endif
}

// ---------------------------------------------------------------------------
// WMMA helpers (wave32, 16x16x32 bf16 -> f32)
// ---------------------------------------------------------------------------
__device__ __forceinline__ v8f zero_v8f() {
    v8f z;
#pragma unroll
    for (int i = 0; i < 8; ++i) z[i] = 0.0f;
    return z;
}

// A fragment: 16(M) x 32(K) bf16, row-major in LDS, row stride `ld` elements.
__device__ __forceinline__ v16bf frag_a(const bf16* base, int ld) {
    const int lane = threadIdx.x & 31;
    const int row  = lane & 15;
    const int hi   = lane >> 4;
    const bf16* r  = base + row * ld + hi * 8;
    union { v16bf v; v8bf h[2]; } u;
    u.h[0] = *reinterpret_cast<const v8bf*>(r);        // K = hi*8 .. +7
    u.h[1] = *reinterpret_cast<const v8bf*>(r + 16);   // K = 16+hi*8 .. +7
    return u.v;
}

// B fragment: 32(K) x 16(N), stored TRANSPOSED in LDS as BT[n][k].
__device__ __forceinline__ v16bf frag_b(const bf16* baseT, int ld) {
    const int lane = threadIdx.x & 31;
    const int n    = lane & 15;
    const int hi   = lane >> 4;
    const bf16* r  = baseT + n * ld + hi * 8;
    union { v16bf v; v8bf h[2]; } u;
    u.h[0] = *reinterpret_cast<const v8bf*>(r);
    u.h[1] = *reinterpret_cast<const v8bf*>(r + 16);
    return u.v;
}

__device__ __forceinline__ v8f wmma_bf16(v16bf a, v16bf b, v8f c) {
    return __builtin_amdgcn_wmma_f32_16x16x32_bf16(false, a, false, b,
                                                   (short)0, c, false, false);
}

// ---------------------------------------------------------------------------
// f32 -> bf16 flat convert (for x)
// ---------------------------------------------------------------------------
__global__ __launch_bounds__(256)
void convert_bf16(const float* __restrict__ src, bf16* __restrict__ dst) {
    const size_t i = ((size_t)blockIdx.x * 256 + threadIdx.x) * 4;
    float4 f = *reinterpret_cast<const float4*>(src + i);
    v4bf h;
    h[0] = (bf16)f.x; h[1] = (bf16)f.y; h[2] = (bf16)f.z; h[3] = (bf16)f.w;
    *reinterpret_cast<v4bf*>(dst + i) = h;
}

// ---------------------------------------------------------------------------
// f32 W[K,N] -> bf16 WT[N,K] (tiled transpose so GEMM B-tiles are row copies)
// ---------------------------------------------------------------------------
__global__ __launch_bounds__(256)
void convert_transpose(const float* __restrict__ W, bf16* __restrict__ WT,
                       int K, int N) {
    __shared__ float t[64][65];
    const int k0 = blockIdx.x * 64;
    const int n0 = blockIdx.y * 64;
#pragma unroll
    for (int i = 0; i < 16; ++i) {
        const int e = threadIdx.x + i * 256;   // 0..4095
        const int r = e >> 6;                  // k in tile
        const int c = e & 63;                  // n in tile
        t[r][c] = W[(size_t)(k0 + r) * N + n0 + c];
    }
    __syncthreads();
#pragma unroll
    for (int i = 0; i < 16; ++i) {
        const int e = threadIdx.x + i * 256;
        const int r = e >> 6;                  // n in tile
        const int c = e & 63;                  // k in tile
        WT[(size_t)(n0 + r) * K + k0 + c] = (bf16)t[c][r];
    }
}

// ---------------------------------------------------------------------------
// GEMM: Y[M,N] = X[M,K](bf16) @ WT[N,K](bf16)^T + bias(f32).
// 128x128x32 tiles, 8 waves, double-buffered async LDS staging.
// ---------------------------------------------------------------------------
template <bool OUT_F32>
__global__ __launch_bounds__(256)
void gemm_bf16(const bf16* __restrict__ X, const bf16* __restrict__ WT,
               const float* __restrict__ bias, void* __restrict__ Yv,
               int M, int N, int K) {
    constexpr int BM = 128, BN = 128, BK = 32;
    constexpr int LDA = BK + 8;   // 40 elems -> rows stay 16B aligned
    constexpr int LDB = BK + 8;

    __shared__ __align__(16) bf16 As[2][BM * LDA];
    __shared__ __align__(16) bf16 BTs[2][BN * LDB];

    const int tid  = threadIdx.x;
    const int wave = tid >> 5;
    const int lane = tid & 31;
    const int n16  = lane & 15;
    const int hi   = lane >> 4;

    const int bm = blockIdx.x * BM;
    const int bn = blockIdx.y * BN;
    const int wm0 = (wave >> 2) * 64;
    const int wn0 = (wave & 3) * 32;

    v8f acc[4][2];
#pragma unroll
    for (int m = 0; m < 4; ++m)
#pragma unroll
        for (int n = 0; n < 2; ++n) acc[m][n] = zero_v8f();

    auto stage = [&](int buf, int kk) {
#pragma unroll
        for (int it = 0; it < 2; ++it) {
            const int e   = tid + it * 256;   // 0..511 chunks of 8 elems
            const int row = e >> 2;           // 128 rows x 4 chunks
            const int c   = (e & 3) * 8;
            cp16(&As[buf][row * LDA + c], X + (size_t)(bm + row) * K + kk + c);
        }
#pragma unroll
        for (int it = 0; it < 2; ++it) {
            const int e   = tid + it * 256;
            const int row = e >> 2;
            const int c   = (e & 3) * 8;
            cp16(&BTs[buf][row * LDB + c], WT + (size_t)(bn + row) * K + kk + c);
        }
    };

    stage(0, 0);
    int cur = 0;
    for (int k0 = 0; k0 < K; k0 += BK) {
        wait_async0();          // copies into `cur` are complete (this wave)
        __syncthreads();        // ... and for all waves
        if (k0 + BK < K) stage(cur ^ 1, k0 + BK);   // prefetch next tile

        v16bf af[4];
#pragma unroll
        for (int m = 0; m < 4; ++m)
            af[m] = frag_a(&As[cur][(wm0 + m * 16) * LDA], LDA);
        v16bf bg[2];
#pragma unroll
        for (int n = 0; n < 2; ++n)
            bg[n] = frag_b(&BTs[cur][(wn0 + n * 16) * LDB], LDB);
#pragma unroll
        for (int m = 0; m < 4; ++m)
#pragma unroll
            for (int n = 0; n < 2; ++n)
                acc[m][n] = wmma_bf16(af[m], bg[n], acc[m][n]);
        cur ^= 1;
    }

    // epilogue: +bias, store
#pragma unroll
    for (int n = 0; n < 2; ++n) {
        const int col = bn + wn0 + n * 16 + n16;
        const float bc = bias[col];
#pragma unroll
        for (int m = 0; m < 4; ++m) {
#pragma unroll
            for (int r = 0; r < 8; ++r) {
                const int row = bm + wm0 + m * 16 + r + 8 * hi;
                const float v = acc[m][n][r] + bc;
                if constexpr (OUT_F32)
                    ((float*)Yv)[(size_t)row * N + col] = v;
                else
                    ((bf16*)Yv)[(size_t)row * N + col] = (bf16)v;
            }
        }
    }
}

// ---------------------------------------------------------------------------
// RMS-norm (full 2048-wide) + interleaved RoPE + head split.
// ---------------------------------------------------------------------------
__global__ __launch_bounds__(256)
void norm_rope(const bf16* __restrict__ qb, const bf16* __restrict__ kb,
               const bf16* __restrict__ vb,
               const float* __restrict__ qn_w, const float* __restrict__ kn_w,
               const float* __restrict__ q_cos, const float* __restrict__ q_sin,
               const float* __restrict__ k_cos, const float* __restrict__ k_sin,
               bf16* __restrict__ qh, bf16* __restrict__ kh,
               bf16* __restrict__ vh) {
    const int bs = blockIdx.x;
    const int b  = bs / S_;
    const int s  = bs % S_;
    const int t  = threadIdx.x;

    const bf16* qrow = qb + (size_t)bs * INNER_;
    const bf16* krow = kb + (size_t)bs * INNER_;
    const bf16* vrow = vb + (size_t)bs * INNER_;

    float ql[8], kl[8], vl[8];
    float sq = 0.f, sk = 0.f;
#pragma unroll
    for (int i = 0; i < 8; ++i) {
        const int j = t * 8 + i;
        ql[i] = (float)qrow[j];
        kl[i] = (float)krow[j];
        vl[i] = (float)vrow[j];
        sq += ql[i] * ql[i];
        sk += kl[i] * kl[i];
    }

    __shared__ float redq[256], redk[256];
    redq[t] = sq; redk[t] = sk;
    __syncthreads();
    for (int off = 128; off > 0; off >>= 1) {
        if (t < off) { redq[t] += redq[t + off]; redk[t] += redk[t + off]; }
        __syncthreads();
    }
    const float rq = rsqrtf(redq[0] * (1.0f / INNER_) + EPS_);
    const float rk = rsqrtf(redk[0] * (1.0f / INNER_) + EPS_);

    const int h  = (t * 8) / DH_;
    const int d0 = (t * 8) % DH_;
    const size_t hbase = ((size_t)(b * H_ + h) * S_ + s) * DH_;
    const size_t cbase = (size_t)s * DH_;

#pragma unroll
    for (int p = 0; p < 4; ++p) {
        const int j = t * 8 + 2 * p;
        const int d = d0 + 2 * p;
        {
            const float x1 = ql[2 * p]     * rq * qn_w[j];
            const float x2 = ql[2 * p + 1] * rq * qn_w[j + 1];
            const float c0 = q_cos[cbase + d],     s0 = q_sin[cbase + d];
            const float c1 = q_cos[cbase + d + 1], s1 = q_sin[cbase + d + 1];
            qh[hbase + d]     = (bf16)(x1 * c0 - x2 * s0);
            qh[hbase + d + 1] = (bf16)(x2 * c1 + x1 * s1);
        }
        {
            const float x1 = kl[2 * p]     * rk * kn_w[j];
            const float x2 = kl[2 * p + 1] * rk * kn_w[j + 1];
            const float c0 = k_cos[cbase + d],     s0 = k_sin[cbase + d];
            const float c1 = k_cos[cbase + d + 1], s1 = k_sin[cbase + d + 1];
            kh[hbase + d]     = (bf16)(x1 * c0 - x2 * s0);
            kh[hbase + d + 1] = (bf16)(x2 * c1 + x1 * s1);
        }
        vh[hbase + d]     = (bf16)vl[2 * p];
        vh[hbase + d + 1] = (bf16)vl[2 * p + 1];
    }
}

// ---------------------------------------------------------------------------
// Flash attention: per (b,h), 64-query tile, online softmax over 32 key tiles.
// Q/K staged via async DMA to LDS; V staged with vector loads + LDS transpose.
// ---------------------------------------------------------------------------
__global__ __launch_bounds__(256)
void attn(const bf16* __restrict__ qh, const bf16* __restrict__ kh,
          const bf16* __restrict__ vh, bf16* __restrict__ ob) {
    constexpr int LQ = 72;
    constexpr int LS = 68;
    const float scale = 0.125f;   // 1/sqrt(64)

    __shared__ __align__(16) bf16 Qs[64 * LQ];
    __shared__ __align__(16) bf16 Ks[64 * LQ];    // [kidx][d]  (BT for Q@K^T)
    __shared__ __align__(16) bf16 VTs[64 * LQ];   // [d][kidx]  (BT for P@V)
    __shared__ __align__(16) bf16 Ps[64 * LQ];
    __shared__ float Ss[64 * LS];
    __shared__ float mrow[64], lrow[64], arow[64];

    const int bh = blockIdx.x;
    const int b  = bh / H_;
    const int h  = bh % H_;
    const int q0 = blockIdx.y * 64;

    const int tid  = threadIdx.x;
    const int wave = tid >> 5;
    const int lane = tid & 31;
    const int n16  = lane & 15;
    const int hi   = lane >> 4;

    const int msub = wave >> 1;
    const int nb   = (wave & 1) * 2;

    const size_t head = (size_t)(b * H_ + h) * S_;
    const bf16* Qg = qh + (head + q0) * DH_;
    const bf16* Kg = kh + head * DH_;
    const bf16* Vg = vh + head * DH_;

    // Q tile: async DMA to LDS
#pragma unroll
    for (int it = 0; it < 2; ++it) {
        const int e   = tid + it * 256;
        const int row = e >> 3;
        const int c8  = (e & 7) * 8;
        cp16(&Qs[row * LQ + c8], Qg + (size_t)row * DH_ + c8);
    }
    if (tid < 64) { mrow[tid] = -1e30f; lrow[tid] = 0.f; }

    v8f oacc[2] = { zero_v8f(), zero_v8f() };

    for (int kk0 = 0; kk0 < S_; kk0 += 64) {
        // prefetch next K/V tiles into cache while computing this one
        if (kk0 + 64 < S_) {
            __builtin_prefetch((const void*)(Kg + (size_t)(kk0 + 64) * DH_ + tid * 16), 0, 1);
            __builtin_prefetch((const void*)(Vg + (size_t)(kk0 + 64) * DH_ + tid * 16), 0, 1);
        }
        // K tile: async DMA, row-major
#pragma unroll
        for (int it = 0; it < 2; ++it) {
            const int e   = tid + it * 256;
            const int row = e >> 3;
            const int c8  = (e & 7) * 8;
            cp16(&Ks[row * LQ + c8], Kg + (size_t)(kk0 + row) * DH_ + c8);
        }
        // V tile: vector load + transposed LDS store
#pragma unroll
        for (int it = 0; it < 2; ++it) {
            const int e   = tid + it * 256;
            const int row = e >> 3;           // kidx
            const int c8  = (e & 7) * 8;      // d base
            v8bf vv = *reinterpret_cast<const v8bf*>(
                Vg + (size_t)(kk0 + row) * DH_ + c8);
#pragma unroll
            for (int j = 0; j < 8; ++j)
                VTs[(c8 + j) * LQ + row] = vv[j];
        }
        wait_async0();
        __syncthreads();

        // S = scale * Q @ K^T
        v8f sacc[2] = { zero_v8f(), zero_v8f() };
#pragma unroll
        for (int kki = 0; kki < 64; kki += 32) {
            v16bf a = frag_a(&Qs[(msub * 16) * LQ + kki], LQ);
#pragma unroll
            for (int j = 0; j < 2; ++j) {
                v16bf bg = frag_b(&Ks[((nb + j) * 16) * LQ + kki], LQ);
                sacc[j] = wmma_bf16(a, bg, sacc[j]);
            }
        }
#pragma unroll
        for (int j = 0; j < 2; ++j) {
            const int col = (nb + j) * 16 + n16;
#pragma unroll
            for (int r = 0; r < 8; ++r) {
                const int row = msub * 16 + r + 8 * hi;
                Ss[row * LS + col] = sacc[j][r] * scale;
            }
        }
        __syncthreads();

        // online softmax (one thread per query row)
        if (tid < 64) {
            const float m_old = mrow[tid];
            float mx = m_old;
            for (int c = 0; c < 64; ++c) mx = fmaxf(mx, Ss[tid * LS + c]);
            const float al = __expf(m_old - mx);
            float sum = 0.f;
            for (int c = 0; c < 64; ++c) {
                const float p = __expf(Ss[tid * LS + c] - mx);
                Ps[tid * LQ + c] = (bf16)p;
                sum += p;
            }
            mrow[tid] = mx;
            lrow[tid] = lrow[tid] * al + sum;
            arow[tid] = al;
        }
        __syncthreads();

        // rescale accumulators, O += P @ V
#pragma unroll
        for (int j = 0; j < 2; ++j)
#pragma unroll
            for (int r = 0; r < 8; ++r)
                oacc[j][r] *= arow[msub * 16 + r + 8 * hi];
#pragma unroll
        for (int kki = 0; kki < 64; kki += 32) {
            v16bf a = frag_a(&Ps[(msub * 16) * LQ + kki], LQ);
#pragma unroll
            for (int j = 0; j < 2; ++j) {
                v16bf bg = frag_b(&VTs[((nb + j) * 16) * LQ + kki], LQ);
                oacc[j] = wmma_bf16(a, bg, oacc[j]);
            }
        }
        __syncthreads();
    }

    // normalize and store: ob[b, s, h*64 + d]
#pragma unroll
    for (int j = 0; j < 2; ++j) {
        const int col = (nb + j) * 16 + n16;
#pragma unroll
        for (int r = 0; r < 8; ++r) {
            const int row = msub * 16 + r + 8 * hi;
            const float v = oacc[j][r] / lrow[row];
            ob[((size_t)b * S_ + q0 + row) * INNER_ + h * DH_ + col] = (bf16)v;
        }
    }
}

// ---------------------------------------------------------------------------
// Host-side orchestration
// ---------------------------------------------------------------------------
extern "C" void kernel_launch(void* const* d_in, const int* in_sizes, int n_in,
                              void* d_out, int out_size, void* d_ws,
                              size_t ws_size, hipStream_t stream) {
    (void)in_sizes; (void)n_in; (void)out_size; (void)ws_size;

    const float* x     = (const float*)d_in[0];
    const float* q_cos = (const float*)d_in[1];
    const float* q_sin = (const float*)d_in[2];
    const float* k_cos = (const float*)d_in[3];
    const float* k_sin = (const float*)d_in[4];
    const float* Wq    = (const float*)d_in[5];
    const float* bq    = (const float*)d_in[6];
    const float* Wk    = (const float*)d_in[7];
    const float* bk    = (const float*)d_in[8];
    const float* Wv    = (const float*)d_in[9];
    const float* bv    = (const float*)d_in[10];
    const float* qn_w  = (const float*)d_in[11];
    const float* kn_w  = (const float*)d_in[12];
    const float* Wo    = (const float*)d_in[13];
    const float* bo    = (const float*)d_in[14];

    const size_t nX = (size_t)B_ * S_ * INNER_;   // 8,388,608
    const size_t nW = (size_t)D_ * INNER_;        // 4,194,304

    bf16* xb  = (bf16*)d_ws;
    bf16* wtq = xb + nX;
    bf16* wtk = wtq + nW;
    bf16* wtv = wtk + nW;
    bf16* wto = wtv + nW;
    bf16* qb  = wto + nW;
    bf16* kb  = qb + nX;
    bf16* vb  = kb + nX;
    bf16* qh  = vb + nX;
    bf16* kh  = qh + nX;
    bf16* vh  = kh + nX;
    bf16* ob  = vh + nX;

    const int M = B_ * S_;   // 4096
    const int N = INNER_;    // 2048
    const int K = D_;        // 2048

    convert_bf16<<<(int)(nX / (256 * 4)), 256, 0, stream>>>(x, xb);
    dim3 gt(K / 64, N / 64);
    convert_transpose<<<gt, 256, 0, stream>>>(Wq, wtq, K, N);
    convert_transpose<<<gt, 256, 0, stream>>>(Wk, wtk, K, N);
    convert_transpose<<<gt, 256, 0, stream>>>(Wv, wtv, K, N);
    convert_transpose<<<gt, 256, 0, stream>>>(Wo, wto, K, N);

    dim3 gg(M / 128, N / 128);
    gemm_bf16<false><<<gg, 256, 0, stream>>>(xb, wtq, bq, qb, M, N, K);
    gemm_bf16<false><<<gg, 256, 0, stream>>>(xb, wtk, bk, kb, M, N, K);
    gemm_bf16<false><<<gg, 256, 0, stream>>>(xb, wtv, bv, vb, M, N, K);

    norm_rope<<<B_ * S_, 256, 0, stream>>>(qb, kb, vb, qn_w, kn_w,
                                           q_cos, q_sin, k_cos, k_sin,
                                           qh, kh, vh);

    attn<<<dim3(B_ * H_, S_ / 64), 256, 0, stream>>>(qh, kh, vh, ob);

    gemm_bf16<true><<<gg, 256, 0, stream>>>(ob, wto, bo, (float*)d_out,
                                            M, INNER_, INNER_);
}
